// PGCU_83064667505035
// MI455X (gfx1250) — matrix operand order repaired and checked by reading, not compiled, not run on hardware
//
#include <hip/hip_runtime.h>

typedef __attribute__((ext_vector_type(16))) _Float16 v16h;
typedef __attribute__((ext_vector_type(8)))  float    v8f;

union F16Frag { uint4 q[2]; v16h v; };
union H8      { uint4 q; _Float16 h[8]; };

#define NPIX (256*256)

__device__ inline v8f vbc(float x) { v8f v; _Pragma("unroll") for (int i = 0; i < 8; ++i) v[i] = x; return v; }
__device__ inline v8f v8zero() { return vbc(0.f); }

// reduce-add across the 16 lanes of each half-wave; result broadcast to all 16
__device__ inline v8f red16(v8f t) {
#pragma unroll
  for (int m = 1; m <= 8; m <<= 1) {
    v8f o;
#pragma unroll
    for (int r = 0; r < 8; ++r) o[r] = __shfl_xor(t[r], m, 32);
    t = t + o;
  }
  return t;
}

__device__ inline v8f red16max(v8f t) {
#pragma unroll
  for (int m = 1; m <= 8; m <<= 1) {
#pragma unroll
    for (int r = 0; r < 8; ++r) t[r] = fmaxf(t[r], __shfl_xor(t[r], m, 32));
  }
  return t;
}

// ---------------------------------------------------------------------------
// generic 3x3 conv, pad=1, arbitrary stride; output channel placement via coTot/coOff
// ---------------------------------------------------------------------------
__global__ void conv3_k(const float* __restrict__ in, const float* __restrict__ w,
                        const float* __restrict__ bias, float* __restrict__ out,
                        int B, int Ci, int Co, int Hin, int Win, int stride,
                        int coTot, int coOff) {
  int Ho = (Hin - 1) / stride + 1;
  int Wo = (Win - 1) / stride + 1;
  int total = B * Co * Ho * Wo;
  int i = blockIdx.x * blockDim.x + threadIdx.x;
  if (i >= total) return;
  int xo = i % Wo; int t = i / Wo;
  int yo = t % Ho; t /= Ho;
  int co = t % Co; int b = t / Co;
  float acc = bias[co];
  int yi0 = yo * stride - 1, xi0 = xo * stride - 1;
  for (int ci = 0; ci < Ci; ++ci)
    for (int dy = 0; dy < 3; ++dy) {
      int yy = yi0 + dy; if (yy < 0 || yy >= Hin) continue;
      for (int dx = 0; dx < 3; ++dx) {
        int xx = xi0 + dx; if (xx < 0 || xx >= Win) continue;
        acc += in[(((size_t)b * Ci + ci) * Hin + yy) * Win + xx] *
               w[((co * Ci + ci) * 3 + dy) * 3 + dx];
      }
    }
  out[(((size_t)b * coTot + coOff + co) * Ho + yo) * Wo + xo] = acc;
}

__global__ void pool2_k(const float* __restrict__ in, float* __restrict__ out,
                        int B, int C, int Hin, int Win, int cTot, int cOff) {
  int Ho = Hin >> 1, Wo = Win >> 1;
  int total = B * C * Ho * Wo;
  int i = blockIdx.x * blockDim.x + threadIdx.x;
  if (i >= total) return;
  int xo = i % Wo; int t = i / Wo;
  int yo = t % Ho; t /= Ho;
  int c = t % C; int b = t / C;
  const float* p = in + (((size_t)b * C + c) * Hin + yo * 2) * Win + xo * 2;
  float m = fmaxf(fmaxf(p[0], p[1]), fmaxf(p[Win], p[Win + 1]));
  out[(((size_t)b * cTot + cOff + c) * Ho + yo) * Wo + xo] = m;
}

// ---------------------------------------------------------------------------
// pack WMMA B-matrices as f16 [N][K] row-major:
//   WF [128][96]: F conv weights, k = tap*8 + ch (ch 0-3 Fm, 4-7 Fq), zero pad k>=72
//   WP [128][128]: projection, n = c*32+k_out, K = d
// ---------------------------------------------------------------------------
__global__ void pack_w_k(const float* __restrict__ Fw, const float* __restrict__ Flw,
                         _Float16* __restrict__ WF, _Float16* __restrict__ WP) {
  int i = blockIdx.x * blockDim.x + threadIdx.x;
  if (i < 128 * 96) {
    int n = i / 96, k = i % 96;
    int t = k >> 3, c = k & 7;
    float v = (t < 9) ? Fw[(n * 8 + c) * 9 + t] : 0.f;
    WF[i] = (_Float16)v;
  } else if (i < 128 * 96 + 128 * 128) {
    int ii = i - 128 * 96;
    int n = ii / 128, d = ii % 128;
    WP[ii] = (_Float16)Flw[(((n >> 5) * 128) + d) * 32 + (n & 31)];
  }
}

// ---------------------------------------------------------------------------
// FmFq producer: Fm = conv3x3(nearest4x(x), FM), Fq = conv3x3(guide, FP)
// stored NHWC8 f16: [B][256][256][8]  (one uint4 per pixel)
// ---------------------------------------------------------------------------
__global__ void fmfq_k(const float* __restrict__ guide, const float* __restrict__ x,
                       const float* __restrict__ FPw, const float* __restrict__ FPb,
                       const float* __restrict__ FMw, const float* __restrict__ FMb,
                       _Float16* __restrict__ out) {
  int i = blockIdx.x * blockDim.x + threadIdx.x;
  if (i >= 8 * NPIX) return;
  int gx = i & 255, gy = (i >> 8) & 255, b = i >> 16;
  float fm[4], fq[4];
#pragma unroll
  for (int co = 0; co < 4; ++co) { fm[co] = FMb[co]; fq[co] = FPb[co]; }
  for (int dy = 0; dy < 3; ++dy) {
    int yy = gy - 1 + dy; if (yy < 0 || yy >= 256) continue;
    for (int dx = 0; dx < 3; ++dx) {
      int xx = gx - 1 + dx; if (xx < 0 || xx >= 256) continue;
      int t = dy * 3 + dx;
      float gv = guide[(size_t)b * NPIX + yy * 256 + xx];
#pragma unroll
      for (int co = 0; co < 4; ++co) fq[co] += gv * FPw[co * 9 + t];
#pragma unroll
      for (int ci = 0; ci < 4; ++ci) {
        float xv = x[(((size_t)b * 4 + ci) * 64 + (yy >> 2)) * 64 + (xx >> 2)];
#pragma unroll
        for (int co = 0; co < 4; ++co) fm[co] += xv * FMw[(co * 4 + ci) * 9 + t];
      }
    }
  }
  H8 o;
#pragma unroll
  for (int co = 0; co < 4; ++co) { o.h[co] = (_Float16)fm[co]; o.h[4 + co] = (_Float16)fq[co]; }
  reinterpret_cast<uint4*>(out)[i] = o.q;
}

// ---------------------------------------------------------------------------
// low-res keys: FVF = LN(Gf . Gl_w + Gl_b), Vf  — one thread per (b, m, c)
// ---------------------------------------------------------------------------
__global__ void keys_k(const float* __restrict__ G, const float* __restrict__ V,
                       const float* __restrict__ Glw, const float* __restrict__ Glb,
                       const float* __restrict__ Glg, const float* __restrict__ Glbe,
                       float* __restrict__ FVFg, float* __restrict__ Vfg) {
  int i = blockIdx.x * blockDim.x + threadIdx.x;
  if (i >= 8 * 16 * 4) return;
  int c = i & 3, m = (i >> 2) & 15, b = i >> 6;
  int y = m >> 2, xq = m & 3;
  float v[32];
#pragma unroll
  for (int k = 0; k < 32; ++k) v[k] = Glb[c * 32 + k];
  for (int d = 0; d < 128; ++d) {
    float gv = G[(((size_t)b * 128 + d) * 4 + y) * 4 + xq];
#pragma unroll
    for (int k = 0; k < 32; ++k) v[k] += gv * Glw[((size_t)c * 128 + d) * 32 + k];
  }
  float mu = 0.f;
#pragma unroll
  for (int k = 0; k < 32; ++k) mu += v[k];
  mu *= (1.f / 32.f);
  float var = 0.f;
#pragma unroll
  for (int k = 0; k < 32; ++k) { float d0 = v[k] - mu; var += d0 * d0; }
  var *= (1.f / 32.f);
  float rs = rsqrtf(var + 1e-5f);
#pragma unroll
  for (int k = 0; k < 32; ++k)
    FVFg[(((size_t)b * 16 + m) * 4 + c) * 32 + k] =
        (v[k] - mu) * rs * Glg[c * 32 + k] + Glbe[c * 32 + k];
  Vfg[((size_t)b * 16 + m) * 4 + c] = V[(((size_t)b * 4 + c) * 4 + y) * 4 + xq];
}

// ---------------------------------------------------------------------------
// fused:  conv(8->128) [WMMA] -> proj(128->128) [WMMA] -> LN -> scores [WMMA]
//         -> softmax over 16 keys -> V reduction
// tile: 16 (x) * 8 (y) pixels, 128 threads = 4 waves; wave w owns rows 2w, 2w+1,
// processed one row-tile at a time to keep live accumulators at 8 tiles.
// Halo has a zero slot (index 180) so padded taps select an LDS *index*, never a
// pointer/value select (which previously lowered to flat+scratch loads).
// ---------------------------------------------------------------------------
__global__ __launch_bounds__(128) void fused_attn_k(
    const _Float16* __restrict__ FmFq, const _Float16* __restrict__ WF,
    const _Float16* __restrict__ WP, const float* __restrict__ Fb,
    const float* __restrict__ Flb, const float* __restrict__ Flg,
    const float* __restrict__ Flbe, const float* __restrict__ FVFg,
    const float* __restrict__ Vfg, float* __restrict__ attn) {
  __shared__ uint4 sHalo[10 * 18 + 1];               // +1 zero slot at index 180
  __shared__ __align__(16) _Float16 sF[128 * 136];   // F tile, ld=136 halfs (conflict-free)
  __shared__ __align__(16) _Float16 sPV[4 * 16 * 136]; // per-wave normalized PVF row tile
  __shared__ __align__(16) _Float16 sFVFh[2048];     // keys [m][c][k] as f16
  __shared__ float sVf[64];

  const int tid = threadIdx.x;
  const int lane = tid & 31, w = tid >> 5;
  const int h = lane >> 4, j = lane & 15;
  const int tx0 = blockIdx.x * 16, ty0 = blockIdx.y * 8, b = blockIdx.z;

  for (int i = tid; i < 2048; i += 128) sFVFh[i] = (_Float16)FVFg[(size_t)b * 2048 + i];
  if (tid < 64) sVf[tid] = Vfg[(size_t)b * 64 + tid];

  const uint4* src = reinterpret_cast<const uint4*>(FmFq);
  for (int i = tid; i < 181; i += 128) {
    uint4 val; val.x = val.y = val.z = val.w = 0u;
    if (i < 180) {
      int y = i / 18, xx = i % 18;
      int gy = ty0 + y - 1, gx = tx0 + xx - 1;
      if (gy >= 0 && gy < 256 && gx >= 0 && gx < 256)
        val = src[((size_t)b * 256 + gy) * 256 + gx];
    }
    sHalo[i] = val;
  }
  __syncthreads();

  const int ty = 2 * w;  // first image row (in tile) owned by this wave

  // ---- GEMM1: implicit conv, one row-tile at a time (8 live acc tiles) ----
#pragma unroll 1
  for (int rt = 0; rt < 2; ++rt) {
    const int row = ty + rt;
    v8f acc[8];
#pragma unroll
    for (int ct = 0; ct < 8; ++ct) acc[ct] = v8zero();
#pragma unroll
    for (int s = 0; s < 3; ++s) {
      // taps for this k-step: h=0 -> {4s, 4s+2}, h=1 -> {4s+1, 4s+3}; tap>=9 -> zero slot.
      // All tap/3, tap%3 are compile-time constants; only a cndmask on the LDS index.
      const int tA0 = 4 * s,     tA1 = 4 * s + 1;
      const int tB0 = 4 * s + 2, tB1 = 4 * s + 3;
      int iA0 = (tA0 < 9) ? ((row + tA0 / 3) * 18 + j + (tA0 % 3)) : 180;
      int iA1 = (tA1 < 9) ? ((row + tA1 / 3) * 18 + j + (tA1 % 3)) : 180;
      int iB0 = (tB0 < 9) ? ((row + tB0 / 3) * 18 + j + (tB0 % 3)) : 180;
      int iB1 = (tB1 < 9) ? ((row + tB1 / 3) * 18 + j + (tB1 % 3)) : 180;
      F16Frag a;
      a.q[0] = sHalo[h ? iA1 : iA0];
      a.q[1] = sHalo[h ? iB1 : iB0];
#pragma unroll
      for (int ct = 0; ct < 8; ++ct) {
        F16Frag bf;
        const uint4* bp = reinterpret_cast<const uint4*>(WF + (ct * 16 + j) * 96 + 32 * s + 8 * h);
        bf.q[0] = bp[0]; bf.q[1] = bp[2];
        acc[ct] = __builtin_amdgcn_wmma_f32_16x16x32_f16(false, a.v, false, bf.v, (short)0,
                                                         acc[ct], false, false);
      }
    }
    // bias + pack F row to LDS f16 (D layout: lane j = channel col, reg r -> pixel tx=r+8h)
#pragma unroll
    for (int ct = 0; ct < 8; ++ct) {
      float bb = Fb[ct * 16 + j];
#pragma unroll
      for (int r = 0; r < 8; ++r)
        sF[(row * 16 + r + 8 * h) * 136 + ct * 16 + j] = (_Float16)(acc[ct][r] + bb);
    }
  }
  __syncthreads();

  const float scale = 0.17677669529663687f;  // 1/sqrt(32)
  _Float16* myPV = sPV + w * 16 * 136;

  // ---- GEMM2 + LN + scores(WMMA) + softmax, per row-tile ----
#pragma unroll 1
  for (int rt = 0; rt < 2; ++rt) {
    const int row = ty + rt;
    v8f acc[8];
#pragma unroll
    for (int ct = 0; ct < 8; ++ct) acc[ct] = v8zero();
#pragma unroll
    for (int s = 0; s < 4; ++s) {
      F16Frag a;
      int p0 = row * 16 + j;
      a.q[0] = *reinterpret_cast<const uint4*>(sF + p0 * 136 + 32 * s + 8 * h);
      a.q[1] = *reinterpret_cast<const uint4*>(sF + p0 * 136 + 32 * s + 16 + 8 * h);
#pragma unroll
      for (int ct = 0; ct < 8; ++ct) {
        F16Frag bf;
        const uint4* bp = reinterpret_cast<const uint4*>(WP + (ct * 16 + j) * 128 + 32 * s + 8 * h);
        bf.q[0] = bp[0]; bf.q[1] = bp[2];
        acc[ct] = __builtin_amdgcn_wmma_f32_16x16x32_f16(false, a.v, false, bf.v, (short)0,
                                                         acc[ct], false, false);
      }
    }

    // LayerNorm per band over K=32 (16 lanes x 2 col-tiles), prescale by 1/sqrt(K),
    // write normalized PVF to this wave's LDS tile (lane j = k index, reg r -> pixel)
#pragma unroll
    for (int c = 0; c < 4; ++c) {
      v8f u0 = acc[2 * c]     + vbc(Flb[c * 32 + j]);
      v8f u1 = acc[2 * c + 1] + vbc(Flb[c * 32 + 16 + j]);
      v8f mu = red16(u0 + u1) * vbc(1.f / 32.f);
      v8f d0 = u0 - mu, d1 = u1 - mu;
      v8f var = red16(d0 * d0 + d1 * d1) * vbc(1.f / 32.f);
      v8f rs;
#pragma unroll
      for (int r = 0; r < 8; ++r) rs[r] = rsqrtf(var[r] + 1e-5f);
      v8f n0 = (d0 * rs * vbc(Flg[c * 32 + j])      + vbc(Flbe[c * 32 + j]))      * vbc(scale);
      v8f n1 = (d1 * rs * vbc(Flg[c * 32 + 16 + j]) + vbc(Flbe[c * 32 + 16 + j])) * vbc(scale);
#pragma unroll
      for (int r = 0; r < 8; ++r) {
        int px = r + 8 * h;
        myPV[px * 136 + c * 32 + j]      = (_Float16)n0[r];
        myPV[px * 136 + c * 32 + 16 + j] = (_Float16)n1[r];
      }
    }

    // scores = PVF(16px x 32) . FVF[c]^T(32 x 16keys) via WMMA; softmax over lanes(=keys)
#pragma unroll
    for (int c = 0; c < 4; ++c) {
      F16Frag a, bf;
      a.q[0]  = *reinterpret_cast<const uint4*>(myPV + j * 136 + c * 32 + 8 * h);
      a.q[1]  = *reinterpret_cast<const uint4*>(myPV + j * 136 + c * 32 + 16 + 8 * h);
      bf.q[0] = *reinterpret_cast<const uint4*>(sFVFh + (j * 4 + c) * 32 + 8 * h);
      bf.q[1] = *reinterpret_cast<const uint4*>(sFVFh + (j * 4 + c) * 32 + 16 + 8 * h);
      v8f sc = __builtin_amdgcn_wmma_f32_16x16x32_f16(false, a.v, false, bf.v, (short)0,
                                                      v8zero(), false, false);
      v8f mx = red16max(sc);
      v8f p;
#pragma unroll
      for (int r = 0; r < 8; ++r) p[r] = __expf(sc[r] - mx[r]);
      v8f S = red16(p);
      v8f O = red16(p * vbc(sVf[j * 4 + c]));
      if (j == 0) {
        float* op = attn + (((size_t)b * 4 + c) * 256 + ty0 + row) * 256 + tx0 + 8 * h;
#pragma unroll
        for (int r = 0; r < 8; ++r) op[r] = O[r] / S[r];
      }
    }
  }
}

// ---------------------------------------------------------------------------
extern "C" void kernel_launch(void* const* d_in, const int* in_sizes, int n_in,
                              void* d_out, int out_size, void* d_ws, size_t ws_size,
                              hipStream_t stream) {
  (void)in_sizes; (void)n_in; (void)out_size; (void)ws_size;
  const float* guide = (const float*)d_in[0];
  const float* x     = (const float*)d_in[1];
  const float* FPw = (const float*)d_in[2],  *FPb = (const float*)d_in[3];
  const float* FMw = (const float*)d_in[4],  *FMb = (const float*)d_in[5];
  const float* Fw  = (const float*)d_in[6],  *Fb  = (const float*)d_in[7];
  const float* GP0w = (const float*)d_in[8],  *GP0b = (const float*)d_in[9];
  const float* GP1w = (const float*)d_in[10], *GP1b = (const float*)d_in[11];
  const float* GP2w = (const float*)d_in[12], *GP2b = (const float*)d_in[13];
  const float* GM0w = (const float*)d_in[14], *GM0b = (const float*)d_in[15];
  const float* GM1w = (const float*)d_in[16], *GM1b = (const float*)d_in[17];
  const float* Gw   = (const float*)d_in[18], *Gb   = (const float*)d_in[19];
  const float* VP0w = (const float*)d_in[20], *VP0b = (const float*)d_in[21];
  const float* VP1w = (const float*)d_in[22], *VP1b = (const float*)d_in[23];
  const float* VP2w = (const float*)d_in[24], *VP2b = (const float*)d_in[25];
  const float* VM0w = (const float*)d_in[26], *VM0b = (const float*)d_in[27];
  const float* VM1w = (const float*)d_in[28], *VM1b = (const float*)d_in[29];
  const float* Vw   = (const float*)d_in[30], *Vb   = (const float*)d_in[31];
  const float* FAw  = (const float*)d_in[32], *FAb  = (const float*)d_in[33];
  const float* Flw  = (const float*)d_in[34], *Flb  = (const float*)d_in[35];
  const float* Flg  = (const float*)d_in[36], *Flbe = (const float*)d_in[37];
  const float* Glw  = (const float*)d_in[38], *Glb  = (const float*)d_in[39];
  const float* Glg  = (const float*)d_in[40], *Glbe = (const float*)d_in[41];

  char* ws = (char*)d_ws;
  size_t off = 0;
  auto carve = [&](size_t bytes) -> void* {
    void* p = ws + off;
    off = (off + bytes + 255) & ~(size_t)255;
    return p;
  };
  _Float16* FmFq = (_Float16*)carve((size_t)8 * NPIX * 8 * 2);   // 8 MB
  _Float16* WF   = (_Float16*)carve(128 * 96 * 2);
  _Float16* WP   = (_Float16*)carve(128 * 128 * 2);
  float* P    = (float*)carve((size_t)8 * 4 * 128 * 128 * 4);    // 2 MB ping
  float* Q    = (float*)carve((size_t)8 * 4 * 128 * 128 * 4);    // 2 MB pong
  float* GV8  = (float*)carve((size_t)8 * 8 * 16 * 4);
  float* VV8  = (float*)carve((size_t)8 * 8 * 16 * 4);
  float* Gbuf = (float*)carve((size_t)8 * 128 * 16 * 4);
  float* Vbuf = (float*)carve((size_t)8 * 4 * 16 * 4);
  float* FVFg = (float*)carve((size_t)8 * 16 * 4 * 32 * 4);
  float* Vfg  = (float*)carve((size_t)8 * 16 * 4 * 4);
  float* attn = (float*)carve((size_t)8 * 4 * NPIX * 4);         // 8 MB

  auto conv = [&](const float* in, const float* w, const float* bias, float* out,
                  int B, int Ci, int Co, int Hin, int stride, int coTot, int coOff) {
    int Ho = (Hin - 1) / stride + 1;
    int total = B * Co * Ho * Ho;
    conv3_k<<<(total + 255) / 256, 256, 0, stream>>>(in, w, bias, out, B, Ci, Co, Hin, Hin,
                                                     stride, coTot, coOff);
  };
  auto pool = [&](const float* in, float* out, int B, int C, int Hin, int cTot, int cOff) {
    int Ho = Hin / 2;
    int total = B * C * Ho * Ho;
    pool2_k<<<(total + 255) / 256, 256, 0, stream>>>(in, out, B, C, Hin, Hin, cTot, cOff);
  };

  // ---- low-res G branch -> GV8 (ch0-3 Gm, ch4-7 Gp) ----
  conv(x, GM0w, GM0b, P, 8, 4, 4, 64, 2, 4, 0);   pool(P, Q, 8, 4, 32, 4, 0);
  conv(Q, GM1w, GM1b, P, 8, 4, 4, 16, 2, 4, 0);   pool(P, GV8, 8, 4, 8, 8, 0);
  conv(guide, GP0w, GP0b, P, 8, 1, 4, 256, 2, 4, 0); pool(P, Q, 8, 4, 128, 4, 0);
  conv(Q, GP1w, GP1b, P, 8, 4, 4, 64, 2, 4, 0);   pool(P, Q, 8, 4, 32, 4, 0);
  conv(Q, GP2w, GP2b, P, 8, 4, 4, 16, 2, 4, 0);   pool(P, GV8, 8, 4, 8, 8, 4);
  conv(GV8, Gw, Gb, Gbuf, 8, 8, 128, 4, 1, 128, 0);

  // ---- low-res V branch -> VV8 (ch0-3 Vm, ch4-7 Vp) ----
  conv(x, VM0w, VM0b, P, 8, 4, 4, 64, 2, 4, 0);   pool(P, Q, 8, 4, 32, 4, 0);
  conv(Q, VM1w, VM1b, P, 8, 4, 4, 16, 2, 4, 0);   pool(P, VV8, 8, 4, 8, 8, 0);
  conv(guide, VP0w, VP0b, P, 8, 1, 4, 256, 2, 4, 0); pool(P, Q, 8, 4, 128, 4, 0);
  conv(Q, VP1w, VP1b, P, 8, 4, 4, 64, 2, 4, 0);   pool(P, Q, 8, 4, 32, 4, 0);
  conv(Q, VP2w, VP2b, P, 8, 4, 4, 16, 2, 4, 0);   pool(P, VV8, 8, 4, 8, 8, 4);
  conv(VV8, Vw, Vb, Vbuf, 8, 8, 4, 4, 1, 4, 0);

  // ---- keys: FVF (LN'd) + Vf ----
  keys_k<<<8, 64, 0, stream>>>(Gbuf, Vbuf, Glw, Glb, Glg, Glbe, FVFg, Vfg);

  // ---- heavy path ----
  pack_w_k<<<(128 * 96 + 128 * 128 + 255) / 256, 256, 0, stream>>>(Fw, Flw, WF, WP);
  fmfq_k<<<(8 * NPIX + 255) / 256, 256, 0, stream>>>(guide, x, FPw, FPb, FMw, FMb, FmFq);
  fused_attn_k<<<dim3(16, 32, 8), 128, 0, stream>>>(FmFq, WF, WP, Fb, Flb, Flg, Flbe,
                                                    FVFg, Vfg, attn);

  // ---- final FA conv -> output ----
  conv(attn, FAw, FAb, (float*)d_out, 8, 4, 4, 256, 1, 4, 0);
}
